// MSD_computer_68367289417756
// MI455X (gfx1250) — matrix operand order repaired and checked by reading, not compiled, not run on hardware
//
#include <hip/hip_runtime.h>
#include <hip/hip_bf16.h>

// ---- CDNA5 WMMA vector types (per bridge doc) ----
typedef __attribute__((ext_vector_type(16))) __bf16 v16bf;
typedef __attribute__((ext_vector_type(8)))  __bf16 v8bf;
typedef __attribute__((ext_vector_type(4)))  __bf16 v4bf;
typedef __attribute__((ext_vector_type(8)))  float  v8f;
typedef __attribute__((ext_vector_type(4)))  float  f4;

#define TT     4096          // time steps
#define SS     6144          // spatial elements per step (2048*3)
#define NT0    3072          // TT - 1024
#define NLAG   1024
#define NROWP  96            // row-tile PAIRS: i0 = 32*bp, covers rows 0..3071
#define GY     4             // grid.y for gram kernel

// ---------------------------------------------------------------
// Kernel 1: row squared norms  r[t] = sum_s x[t,s]^2   (fp32 source)
// ---------------------------------------------------------------
__global__ __launch_bounds__(256)
void msd_rownorm_kernel(const float* __restrict__ x, float* __restrict__ r) {
    const int row = blockIdx.x;
    const float* p = x + (size_t)row * SS;
    float acc = 0.f;
    for (int i = threadIdx.x; i < SS; i += 256) {
        float v = p[i];
        acc += v * v;
    }
    __shared__ float sm[256];
    sm[threadIdx.x] = acc;
    __syncthreads();
    for (int s = 128; s > 0; s >>= 1) {
        if (threadIdx.x < s) sm[threadIdx.x] += sm[threadIdx.x + s];
        __syncthreads();
    }
    if (threadIdx.x == 0) r[row] = sm[0];
}

// ---------------------------------------------------------------
// Kernel 1b: zero helper (atomic fallback path only)
// ---------------------------------------------------------------
__global__ void msd_zero_kernel(float* __restrict__ p, int n) {
    int i = blockIdx.x * blockDim.x + threadIdx.x;
    if (i < n) p[i] = 0.f;
}

// ---------------------------------------------------------------
// Kernel 1c: one-time fp32 -> bf16 staging copy (row-major, same layout)
// ---------------------------------------------------------------
__global__ __launch_bounds__(256)
void msd_cvt_kernel(const float* __restrict__ x, __bf16* __restrict__ xb) {
    size_t i = ((size_t)blockIdx.x * 256 + threadIdx.x) * 4;   // grid sized exactly
    f4 v = *(const f4*)(x + i);
    v4bf o;
    o[0] = (__bf16)v[0]; o[1] = (__bf16)v[1];
    o[2] = (__bf16)v[2]; o[3] = (__bf16)v[3];
    *(v4bf*)(xb + i) = o;
}

// ---------------------------------------------------------------
// Kernel 2: banded Gram via bf16 WMMA, 2x2 register blocking.
// Wave computes 32x32 patch: row tiles (i0,i0+16) x col tiles (j0,j0+16),
// j0 = i0 + 16*dbase, dbase even in [0,64]. Lag offsets of the 4 tiles:
// {dbase, dbase+1, dbase-1, dbase}; td-in-[0,1024) mask => exact coverage.
// STAGED=true : operands loaded directly as bf16 (2 b128 loads / WMMA).
// STAGED=false: fp32 loads + v_cvt_pk_bf16_f32   (4 b128 loads / WMMA).
// ---------------------------------------------------------------
__device__ __forceinline__ void pack4(v16bf& d, int base, f4 v) {
    d[base + 0] = (__bf16)v[0];
    d[base + 1] = (__bf16)v[1];
    d[base + 2] = (__bf16)v[2];
    d[base + 3] = (__bf16)v[3];
}

__device__ __forceinline__ v16bf join16(v8bf lo, v8bf hi) {
    return __builtin_shufflevector(lo, hi, 0, 1, 2, 3, 4, 5, 6, 7,
                                           8, 9, 10, 11, 12, 13, 14, 15);
}

__device__ __forceinline__ void scatter_tile(float* Cl, const v8f& acc,
                                             int d16, int l15, int half) {
    // C/D layout: VGPR v, lanes 0-15 -> (M=v, N=lane); lanes 16-31 -> (M=v+8, N=lane-16)
    #pragma unroll
    for (int v = 0; v < 8; ++v) {
        const int M  = v + 8 * half;
        const int td = d16 + l15 - M;
        if (td >= 0 && td < NLAG) atomicAdd(&Cl[td], acc[v]);
    }
}

template <bool STAGED>
__global__ __launch_bounds__(128)
void msd_gram_band_kernel(const float* __restrict__ x,
                          const __bf16* __restrict__ xb,
                          float* __restrict__ partials,
                          int atomicMode) {
    __shared__ float Cl[NLAG];
    for (int t = threadIdx.x; t < NLAG; t += 128) Cl[t] = 0.f;
    __syncthreads();

    const int i0   = blockIdx.x * 32;      // row-tile pair base
    const int by   = blockIdx.y;           // 0..3
    const int lane = threadIdx.x & 31;
    const int wave = threadIdx.x >> 5;     // 0..3
    const int half = lane >> 4;
    const int l15  = lane & 15;

    // A operand (16-bit A 16x32): lanes 0-15 K {k0+0..7, k0+16..23},
    //                             lanes 16-31 K {k0+8..15, k0+24..31}
    const size_t aOff0 = (size_t)(i0 + l15) * SS + 8 * half;
    const size_t aOff1 = aOff0 + (size_t)16 * SS;

    const int slot = by + GY * wave;       // 0..15
    for (int dbase = 2 * slot; dbase <= 64; dbase += 32) {
        const int j0 = i0 + 16 * dbase;
        // B operand (16-bit B 32x16): lanes 0-15 K {k0+0..7, k0+16..23}, lanes 16-31 K {k0+8..15, k0+24..31}
        const size_t bOff0 = (size_t)(j0 + l15) * SS + 8 * half;
        const size_t bOff1 = bOff0 + (size_t)16 * SS;

        v8f acc00 = {}, acc01 = {}, acc10 = {}, acc11 = {};
        for (int k0 = 0; k0 < SS; k0 += 32) {
            v16bf av0, av1, bv0, bv1;
            if constexpr (STAGED) {
                const __bf16* a0p = xb + aOff0 + k0;
                const __bf16* a1p = xb + aOff1 + k0;
                const __bf16* b0p = xb + bOff0 + k0;
                const __bf16* b1p = xb + bOff1 + k0;
                av0 = join16(*(const v8bf*)(a0p), *(const v8bf*)(a0p + 16));
                av1 = join16(*(const v8bf*)(a1p), *(const v8bf*)(a1p + 16));
                bv0 = join16(*(const v8bf*)(b0p), *(const v8bf*)(b0p + 16));
                bv1 = join16(*(const v8bf*)(b1p), *(const v8bf*)(b1p + 16));
            } else {
                const float* a0p = x + aOff0 + k0;
                const float* a1p = x + aOff1 + k0;
                const float* b0p = x + bOff0 + k0;
                const float* b1p = x + bOff1 + k0;
                pack4(av0, 0, *(const f4*)(a0p));      pack4(av0, 4, *(const f4*)(a0p + 4));
                pack4(av0, 8, *(const f4*)(a0p + 16)); pack4(av0, 12, *(const f4*)(a0p + 20));
                pack4(av1, 0, *(const f4*)(a1p));      pack4(av1, 4, *(const f4*)(a1p + 4));
                pack4(av1, 8, *(const f4*)(a1p + 16)); pack4(av1, 12, *(const f4*)(a1p + 20));
                pack4(bv0, 0, *(const f4*)(b0p));      pack4(bv0, 4, *(const f4*)(b0p + 4));
                pack4(bv0, 8, *(const f4*)(b0p + 16)); pack4(bv0, 12, *(const f4*)(b0p + 20));
                pack4(bv1, 0, *(const f4*)(b1p));      pack4(bv1, 4, *(const f4*)(b1p + 4));
                pack4(bv1, 8, *(const f4*)(b1p + 16)); pack4(bv1, 12, *(const f4*)(b1p + 20));
            }

            // 4 independent WMMAs sharing the 2+2 operand blocks
            acc00 = __builtin_amdgcn_wmma_f32_16x16x32_bf16(
                false, av0, false, bv0, (short)0, acc00, false, false);
            acc01 = __builtin_amdgcn_wmma_f32_16x16x32_bf16(
                false, av0, false, bv1, (short)0, acc01, false, false);
            acc10 = __builtin_amdgcn_wmma_f32_16x16x32_bf16(
                false, av1, false, bv0, (short)0, acc10, false, false);
            acc11 = __builtin_amdgcn_wmma_f32_16x16x32_bf16(
                false, av1, false, bv1, (short)0, acc11, false, false);
        }

        scatter_tile(Cl, acc00, 16 * dbase,       l15, half);
        scatter_tile(Cl, acc01, 16 * (dbase + 1), l15, half);
        scatter_tile(Cl, acc10, 16 * (dbase - 1), l15, half);
        scatter_tile(Cl, acc11, 16 * dbase,       l15, half);
    }
    __syncthreads();

    if (atomicMode) {
        for (int t = threadIdx.x; t < NLAG; t += 128)
            atomicAdd(&partials[t], Cl[t]);
    } else {
        float* dst = partials + (size_t)(blockIdx.x * GY + by) * NLAG;
        for (int t = threadIdx.x; t < NLAG; t += 128)
            dst[t] = Cl[t];
    }
}

// ---------------------------------------------------------------
// Kernel 3: msd[td] = (SqA(td) + SqB - 2*C(td)) / (NT0*SS), fp64 reduce
// ---------------------------------------------------------------
__global__ __launch_bounds__(256)
void msd_finalize_kernel(const float* __restrict__ r,
                         const float* __restrict__ partials, int npartials,
                         float* __restrict__ out) {
    const int td = blockIdx.x;
    double a = 0.0;
    for (int t = threadIdx.x; t < NT0; t += 256)
        a += (double)r[t + td] + (double)r[t];
    double c = 0.0;
    for (int p = threadIdx.x; p < npartials; p += 256)
        c += (double)partials[(size_t)p * NLAG + td];

    __shared__ double sm[256];
    sm[threadIdx.x] = a - 2.0 * c;
    __syncthreads();
    for (int s = 128; s > 0; s >>= 1) {
        if (threadIdx.x < s) sm[threadIdx.x] += sm[threadIdx.x + s];
        __syncthreads();
    }
    if (threadIdx.x == 0)
        out[td] = (float)(sm[0] / ((double)NT0 * (double)SS));
}

// ---------------------------------------------------------------
extern "C" void kernel_launch(void* const* d_in, const int* in_sizes, int n_in,
                              void* d_out, int out_size, void* d_ws, size_t ws_size,
                              hipStream_t stream) {
    const float* x = (const float*)d_in[0];
    float* out = (float*)d_out;
    char*  ws  = (char*)d_ws;

    const size_t NPART_FULL = (size_t)NROWP * GY;                    // 384 slices
    const size_t partBytes  = NPART_FULL * NLAG * sizeof(float);     // 1.5 MB
    const size_t rBytes     = (size_t)TT * sizeof(float);            // 16 KB
    const size_t xbBytes    = (size_t)TT * SS * sizeof(__bf16);      // 50.3 MB
    const size_t need_bf16  = xbBytes + partBytes + rBytes;
    const size_t need_part  = partBytes + rBytes;

    if (ws_size >= need_bf16) {
        // Preferred: bf16-staged operands (L2-resident, 32 B/lane/WMMA)
        __bf16* xb       = (__bf16*)ws;
        float*  partials = (float*)(ws + xbBytes);
        float*  r        = (float*)(ws + xbBytes + partBytes);

        msd_rownorm_kernel<<<TT, 256, 0, stream>>>(x, r);
        msd_cvt_kernel<<<(unsigned)((size_t)TT * SS / 4 / 256), 256, 0, stream>>>(x, xb);
        msd_gram_band_kernel<true><<<dim3(NROWP, GY), 128, 0, stream>>>(x, xb, partials, 0);
        msd_finalize_kernel<<<NLAG, 256, 0, stream>>>(r, partials, (int)NPART_FULL, out);
    } else if (ws_size >= need_part) {
        // fp32 operands converted in-loop, deterministic partial slices
        float* partials = (float*)ws;
        float* r        = (float*)(ws + partBytes);

        msd_rownorm_kernel<<<TT, 256, 0, stream>>>(x, r);
        msd_gram_band_kernel<false><<<dim3(NROWP, GY), 128, 0, stream>>>(x, nullptr, partials, 0);
        msd_finalize_kernel<<<NLAG, 256, 0, stream>>>(r, partials, (int)NPART_FULL, out);
    } else {
        // Minimal-workspace fallback: fp32 path + global f32 atomics (~20 KB)
        float* partials = (float*)ws;
        float* r        = (float*)(ws + NLAG * sizeof(float));

        msd_rownorm_kernel<<<TT, 256, 0, stream>>>(x, r);
        msd_zero_kernel<<<(NLAG + 255) / 256, 256, 0, stream>>>(partials, NLAG);
        msd_gram_band_kernel<false><<<dim3(NROWP, GY), 128, 0, stream>>>(x, nullptr, partials, 1);
        msd_finalize_kernel<<<NLAG, 256, 0, stream>>>(r, partials, 1, out);
    }
}
